// GCNModel_1778116460904
// MI455X (gfx1250) — matrix operand order repaired and checked by reading, not compiled, not run on hardware
//
#include <hip/hip_runtime.h>
#include <hip/hip_bf16.h>
#include <stdint.h>

#define NN 100000
#define NE 600000
#define F  128

typedef __attribute__((ext_vector_type(16))) __bf16 v16bf;
typedef __attribute__((ext_vector_type(8)))  float  v8f;

union FragBF { v16bf v; uint4 q[2]; };

__device__ __forceinline__ unsigned short f2bf(float f) {
    unsigned u = __float_as_uint(f);
    u += 0x7fffu + ((u >> 16) & 1u);          // round-to-nearest-even
    return (unsigned short)(u >> 16);
}

// ---------------- degree / norm ----------------
__global__ void k_deg_init(float* deg, int n) {
    int i = blockIdx.x * blockDim.x + threadIdx.x;
    if (i < n) deg[i] = 1.0f;                 // self-loop weight
}

__global__ void k_deg_accum(const int* __restrict__ dst, const float* __restrict__ w,
                            float* __restrict__ deg, int e) {
    int i = blockIdx.x * blockDim.x + threadIdx.x;
    if (i < e) atomicAdd(&deg[dst[i]], w[i]);
}

__global__ void k_dinv(float* deg, int n) {
    int i = blockIdx.x * blockDim.x + threadIdx.x;
    if (i < n) deg[i] = rsqrtf(deg[i]);       // deg >= 1 always (self loop)
}

// ---------------- f32 -> bf16 convert ----------------
__global__ void k_cvt_bf16(const float* __restrict__ x, unsigned short* __restrict__ o, int n4) {
    int i = blockIdx.x * blockDim.x + threadIdx.x;
    if (i >= n4) return;
    float4 v = reinterpret_cast<const float4*>(x)[i];
    ushort4 p;
    p.x = f2bf(v.x); p.y = f2bf(v.y); p.z = f2bf(v.z); p.w = f2bf(v.w);
    reinterpret_cast<ushort4*>(o)[i] = p;
}

// ---------------- WMMA bf16 GEMM: C[M,128] = A[M,128](bf16) @ W[128,128](f32->bf16) (+bias) ----
#define SB_STRIDE 136   // bf16 elements per LDS column (272B: 16B aligned, conflict-free b128 reads)

__global__ __launch_bounds__(256, 1)
void k_gemm_bf16(const unsigned short* __restrict__ A, const float* __restrict__ Wg,
                 const float* __restrict__ bias, float* __restrict__ C, int M) {
    __shared__ unsigned short sB[F * SB_STRIDE];

    int tid = threadIdx.x;
    // Stage W (row-major [k][n], f32) into LDS column-major bf16: sB[n*SB_STRIDE + k]
    for (int idx = tid; idx < F * F; idx += 256) {
        int k = idx >> 7, n = idx & 127;
        sB[n * SB_STRIDE + k] = f2bf(Wg[idx]);
    }
    __syncthreads();

    int wv   = tid >> 5;
    int lane = tid & 31;
    int half = lane >> 4;        // 0: lanes 0-15, 1: lanes 16-31
    int l16  = lane & 15;
    int mBase = blockIdx.x * 128 + wv * 16;
    int rowA  = mBase + l16;
    int rowCl = rowA < M ? rowA : (M - 1);    // clamp A loads for the M tail

    const uint4* paBase = reinterpret_cast<const uint4*>(A + (size_t)rowCl * F);

    v8f acc[8];
    v8f zero = {0.f, 0.f, 0.f, 0.f, 0.f, 0.f, 0.f, 0.f};
#pragma unroll
    for (int nt = 0; nt < 8; ++nt) acc[nt] = zero;

#pragma unroll
    for (int kt = 0; kt < 4; ++kt) {
        // A fragment: K = kt*32 + half*16 .. +15 (contiguous)
        FragBF a;
        const uint4* pa = paBase + kt * 4 + half * 2;   // uint4 = 8 bf16
        a.q[0] = pa[0];
        a.q[1] = pa[1];
#pragma unroll
        for (int nt = 0; nt < 8; ++nt) {
            // B fragment: col n = nt*16+l16; K = kt*32 + half*16 .. +15 (contiguous)
            FragBF b;
            const uint4* pb = reinterpret_cast<const uint4*>(
                &sB[(nt * 16 + l16) * SB_STRIDE + kt * 32 + half * 16]);
            b.q[0] = pb[0];
            b.q[1] = pb[1];
            acc[nt] = __builtin_amdgcn_wmma_f32_16x16x32_bf16(
                false, a.v, false, b.v, (short)0, acc[nt], false, false);
        }
    }

    // C tile layout: VGPR r -> M = r (lanes 0-15) / 8+r (lanes 16-31), N = l16
#pragma unroll
    for (int nt = 0; nt < 8; ++nt) {
        int col = nt * 16 + l16;
        float bv = bias ? bias[col] : 0.0f;
#pragma unroll
        for (int r = 0; r < 8; ++r) {
            int row = mBase + half * 8 + r;
            if (row < M) C[(size_t)row * F + col] = acc[nt][r] + bv;
        }
    }
}

// ---------------- edge scatter: one wave32 per edge ----------------
__global__ void k_scatter(const int* __restrict__ src, const int* __restrict__ dst,
                          const float* __restrict__ w, const float* __restrict__ dinv,
                          const float* __restrict__ h, float* __restrict__ agg, int E) {
    int gw   = (blockIdx.x * blockDim.x + threadIdx.x) >> 5;
    int lane = threadIdx.x & 31;
    if (gw >= E) return;
    int s = src[gw], d = dst[gw];
    float nrm = dinv[s] * w[gw] * dinv[d];
    float4 v  = reinterpret_cast<const float4*>(h + (size_t)s * F)[lane];
    float* ap = agg + (size_t)d * F + lane * 4;
    atomicAdd(ap + 0, v.x * nrm);
    atomicAdd(ap + 1, v.y * nrm);
    atomicAdd(ap + 2, v.z * nrm);
    atomicAdd(ap + 3, v.w * nrm);
}

// ---------------- epilogue: agg + selfloop + bias (+relu) -> bf16 ----------------
__global__ void k_post(const float* __restrict__ agg, const float* __restrict__ h,
                       const float* __restrict__ dinv, const float* __restrict__ bias,
                       unsigned short* __restrict__ bfOut, int relu, int n4) {
    int i = blockIdx.x * blockDim.x + threadIdx.x;
    if (i >= n4) return;
    int node = i >> 5;           // 32 float4 per node row
    int c4   = i & 31;
    float di = dinv[node];
    float sl = di * di;          // self-loop norm: dinv*1*dinv
    float4 a  = reinterpret_cast<const float4*>(agg)[i];
    float4 hh = reinterpret_cast<const float4*>(h)[i];
    float4 b  = reinterpret_cast<const float4*>(bias)[c4];
    float4 t;
    t.x = a.x + hh.x * sl + b.x;
    t.y = a.y + hh.y * sl + b.y;
    t.z = a.z + hh.z * sl + b.z;
    t.w = a.w + hh.w * sl + b.w;
    if (relu) {
        t.x = fmaxf(t.x, 0.f); t.y = fmaxf(t.y, 0.f);
        t.z = fmaxf(t.z, 0.f); t.w = fmaxf(t.w, 0.f);
    }
    ushort4 p;
    p.x = f2bf(t.x); p.y = f2bf(t.y); p.z = f2bf(t.z); p.w = f2bf(t.w);
    reinterpret_cast<ushort4*>(bfOut)[i] = p;
}

extern "C" void kernel_launch(void* const* d_in, const int* in_sizes, int n_in,
                              void* d_out, int out_size, void* d_ws, size_t ws_size,
                              hipStream_t stream) {
    const float* x   = (const float*)d_in[0];
    const int*   ei  = (const int*)d_in[1];   // [2, NE]: src row then dst row
    const float* ew  = (const float*)d_in[2];
    const float* W1  = (const float*)d_in[3];
    const float* b1  = (const float*)d_in[4];
    const float* W2  = (const float*)d_in[5];
    const float* b2  = (const float*)d_in[6];
    const float* Wl  = (const float*)d_in[7];
    const float* bl  = (const float*)d_in[8];
    float*       out = (float*)d_out;

    const int* src = ei;
    const int* dst = ei + NE;

    // Scratch carve-out (256B aligned)
    char* ws = (char*)d_ws;
    size_t off = 0;
    auto carve = [&](size_t bytes) {
        void* p = ws + off;
        off = (off + bytes + 255) & ~(size_t)255;
        return p;
    };
    float*          dinv = (float*)carve((size_t)NN * 4);
    unsigned short* bfA  = (unsigned short*)carve((size_t)NN * F * 2);
    float*          h    = (float*)carve((size_t)NN * F * 4);
    float*          agg  = (float*)carve((size_t)NN * F * 4);

    const int BT = 256;
    dim3 blk(BT);
    int gN    = (NN + BT - 1) / BT;
    int gE    = (NE + BT - 1) / BT;
    int gRow4 = (NN * 32 + BT - 1) / BT;      // N*128/4 float4 elements
    int gEw   = (NE * 32 + BT - 1) / BT;      // one wave per edge
    int gGemm = (NN + 127) / 128;             // 782 blocks, 8 waves x 16 rows

    // normalization
    k_deg_init<<<gN, blk, 0, stream>>>(dinv, NN);
    k_deg_accum<<<gE, blk, 0, stream>>>(dst, ew, dinv, NE);
    k_dinv<<<gN, blk, 0, stream>>>(dinv, NN);

    // layer 1
    k_cvt_bf16<<<gRow4, blk, 0, stream>>>(x, bfA, NN * 32);
    k_gemm_bf16<<<gGemm, blk, 0, stream>>>(bfA, W1, nullptr, h, NN);
    hipMemsetAsync(agg, 0, (size_t)NN * F * 4, stream);
    k_scatter<<<gEw, blk, 0, stream>>>(src, dst, ew, dinv, h, agg, NE);
    k_post<<<gRow4, blk, 0, stream>>>(agg, h, dinv, b1, bfA, 1, NN * 32);

    // layer 2
    k_gemm_bf16<<<gGemm, blk, 0, stream>>>(bfA, W2, nullptr, h, NN);
    hipMemsetAsync(agg, 0, (size_t)NN * F * 4, stream);
    k_scatter<<<gEw, blk, 0, stream>>>(src, dst, ew, dinv, h, agg, NE);
    k_post<<<gRow4, blk, 0, stream>>>(agg, h, dinv, b2, bfA, 0, NN * 32);

    // final linear (bias fused)
    k_gemm_bf16<<<gGemm, blk, 0, stream>>>(bfA, Wl, bl, out, NN);
}
